// AZConv2d_24558622998996
// MI455X (gfx1250) — matrix operand order repaired and checked by reading, not compiled, not run on hardware
//
#include <hip/hip_runtime.h>
#include <hip/hip_bf16.h>

// ---------------------------------------------------------------------------
// AZConv2d (anisotropic adaptive gaussian conv) for MI455X / gfx1250.
//
// out_o(l) = pw_b[o] + sum_{r,s} w[r,s](l) * u[o*4+r](neigh_s(l))
//   u = Ueff(256x64) @ x(64xL),   Ueff = fold(pw_w, value_w)
//
// Fully fused tile kernel: per 8x16 spatial tile, compute the 10x18 u-halo
// with v_wmma_f32_16x16x32_f16 straight into LDS (needs CDNA5's 320 KB/WGP
// LDS), compute gating weights w in-kernel, then the 9-point weighted gather
// runs from LDS.  HBM ~= x-halo (24 MB) + out (34 MB) ~= 58 MB -> ~2.5 us.
//
// CDNA5 paths used:
//  * v_wmma_f32_16x16x32_f16 for all dense math (gate/geom + u GEMM)
//  * TENSOR_LOAD_TO_LDS (TDM, 1-D descriptor) to stage the 32 KB folded
//    weight matrix into LDS, tracked with s_wait_tensorcnt
//  * unconditional clamped halo loads + post-load select -> batched clauses
//  * LDS-resident u halo (92 KB) for the 9-point gather (ds_load_b64)
// ---------------------------------------------------------------------------

#define C_IN   64
#define C_OUT  64
#define HH     128
#define WW     128
#define LPIX   16384          // H*W
#define BB     8

#define TH     8              // tile rows   (== waves per block)
#define TW     16             // tile cols   (== WMMA N)
#define HY     (TH + 2)       // 10 halo rows
#define HX     (TW + 2)       // 18 halo cols
#define NHALO  (HY * HX)      // 180 halo pixels
#define NGRP   ((NHALO + 15) / 16)   // 12 WMMA pixel-groups over the halo

typedef __attribute__((ext_vector_type(16))) _Float16 v16h;
typedef __attribute__((ext_vector_type(8)))  float    v8f;
typedef __attribute__((ext_vector_type(4)))  _Float16 v4h;

typedef __attribute__((ext_vector_type(4))) unsigned int u32x4;
typedef __attribute__((ext_vector_type(8))) int          i32x8;
typedef __attribute__((ext_vector_type(4))) int          i32x4;

// workspace layout (bytes) -- only tiny folded weights
#define UEFF_OFF  0u                  // 256*64 f16 = 32 KB
#define WG16_OFF  (64u * 1024u)       // 16*64  f16 =  2 KB

__device__ __forceinline__ float softplus_f(float x) {
    float ax = fabsf(x);
    return fmaxf(x, 0.0f) + __logf(1.0f + __expf(-ax));
}

// K index inside a 32-wide K-step for the 16-bit A-fragment layout
// (lanes hold rows M=lane%16; element t of v16h):
//   K = t + hi*16 (contiguous, matches B fragments)
__device__ __forceinline__ int a_kidx(int t, int hi, int kk) {
    return t + hi * 16 + kk * 32;
}

// ---------------------------------------------------------------------------
// Kernel 0: Ueff[(o,r)][c] = sum_d pw_w[o][r*64+d] * value_w[d][c]  (f16)
//           Wg16 = 16x64 stacked gate(4) + geom(12) weights          (f16)
// ---------------------------------------------------------------------------
__global__ __launch_bounds__(256) void k_prep(
    const float* __restrict__ gate_w, const float* __restrict__ value_w,
    const float* __restrict__ geom_w, const float* __restrict__ pw_w,
    _Float16* __restrict__ ueff, _Float16* __restrict__ wg16) {
    int t = threadIdx.x;          // m = o*4 + r
    int o = t >> 2, r = t & 3;
    const float* pwrow = pw_w + o * 256 + r * 64;
    for (int c = 0; c < 64; ++c) {
        float s = 0.0f;
        #pragma unroll 8
        for (int d = 0; d < 64; ++d) s += pwrow[d] * value_w[d * 64 + c];
        ueff[t * 64 + c] = (_Float16)s;
    }
    if (t < 16) {
        const float* row = (t < 4) ? (gate_w + t * 64) : (geom_w + (t - 4) * 64);
        for (int c = 0; c < 64; ++c) wg16[t * 64 + c] = (_Float16)row[c];
    }
}

// ---------------------------------------------------------------------------
// Fused tile kernel. Block = 256 threads = 8 waves. Tile = 8x16 core pixels.
// LDS: 92160 (sU) + 32768 (sUeff) + 18432 (sW) + 8192 (sG) = 151.5 KB
//      -> 2 workgroups per 320 KB WGP.
// ---------------------------------------------------------------------------
__global__ __launch_bounds__(256) void k_fused(
    const float* __restrict__ x,
    const float* __restrict__ gate_b, const float* __restrict__ geom_b,
    const float* __restrict__ pw_b,
    const _Float16* __restrict__ ueff, const _Float16* __restrict__ wg16,
    float* __restrict__ out) {

    __shared__ _Float16 sU[NHALO * 256];     // u over the halo (f16)
    __shared__ _Float16 sUeff[256 * 64];     // folded weights (f16)
    __shared__ float    sW[TH * TW][36];     // per-core-pixel gating weights
    __shared__ float    sG[8][16][16];       // gate/geom tile transpose

    const int tid  = threadIdx.x;
    const int wave = tid >> 5;               // 0..7 == core row
    const int lane = tid & 31;
    const int col  = lane & 15;              // pixel column / A row
    const int hi   = lane >> 4;

    const int blk = blockIdx.x;              // b * 128 + tile
    const int b   = blk >> 7;
    const int til = blk & 127;
    const int ty0 = (til >> 3) * TH;         // 16 tile-rows
    const int tx0 = (til & 7)  * TW;         //  8 tile-cols
    const float* xb = x + ((long)b << 20);   // x[b][k][pix] at xb[(k<<14)+pix]

    // ---- stage Ueff into LDS with the Tensor Data Mover ---------------------
    // 1-D D#: 8192 x 4B contiguous, global ueff -> LDS sUeff. Issued by wave 0
    // only (TENSORcnt is per-wave); the block barrier releases everyone else.
    if (wave == 0) {
        const unsigned lds_off = (unsigned)(uintptr_t)(void*)sUeff;  // LDS byte offset
        const unsigned long long ga = (unsigned long long)(uintptr_t)ueff;
        u32x4 g0;
        g0[0] = 1u;                                     // count=1, user mode
        g0[1] = lds_off;                                // lds_addr
        g0[2] = (unsigned)(ga & 0xffffffffu);           // global_addr[31:0]
        g0[3] = (unsigned)((ga >> 32) & 0x01ffffffu)    // global_addr[56:32]
                | (2u << 30);                           // type = 2 ("image")
        i32x8 g1;
        g1[0] = 0x00020000;    // workgroup_mask=0, data_size=2 (4 B)
        g1[1] = 0x20000000;    // tensor_dim0[15:0]=8192 in bits[63:48]
        g1[2] = 0x00010000;    // tensor_dim0 hi=0, tensor_dim1=1 in bits[95:80]
        g1[3] = 0x20000000;    // tile_dim0=8192 in bits[127:112]
        g1[4] = 0x00000001;    // tile_dim1=1, tile_dim2=0
        g1[5] = 0x00002000;    // tensor_dim0_stride=8192 (bits[191:160])
        g1[6] = 0;             // stride hi / tensor_dim1_stride lo
        g1[7] = 0;             // tensor_dim1_stride hi
        i32x4 z4 = {0, 0, 0, 0};                        // groups 2/3: unused (<=2D)
        i32x8 z8 = {0, 0, 0, 0, 0, 0, 0, 0};
        __builtin_amdgcn_tensor_load_to_lds(g0, g1, z4, z4, z8, 0);
        __builtin_amdgcn_s_wait_tensorcnt(0);
    }

    // ---- phase A: gate/geom GEMM for this wave's core row -------------------
    {
        const int pixc = (ty0 + wave) * WW + tx0 + col;   // always in-bounds
        v16h bf[2];
        #pragma unroll
        for (int kk = 0; kk < 2; ++kk) {
            const int kbase = kk * 32 + hi * 16;
            #pragma unroll
            for (int t = 0; t < 16; ++t)
                bf[kk][t] = (_Float16)xb[((long)(kbase + t) << 14) + pixc];
        }
        v8f g = {};
        #pragma unroll
        for (int kk = 0; kk < 2; ++kk) {
            v16h a;
            #pragma unroll
            for (int t = 0; t < 16; ++t)
                a[t] = wg16[col * 64 + a_kidx(t, hi, kk)];
            g = __builtin_amdgcn_wmma_f32_16x16x32_f16(false, a, false, bf[kk],
                                                       (short)0, g, false, false);
        }
        #pragma unroll
        for (int t = 0; t < 8; ++t) sG[wave][t + 8 * hi][col] = g[t];
    }

    __syncthreads();   // sUeff staged (TDM complete), sG per-wave stores done

    // ---- phase B: u halo GEMM into LDS (12 groups striped over 8 waves) -----
    for (int g = wave; g < NGRP; g += 8) {
        const int p  = g * 16 + col;             // halo pixel id (may be >=NHALO)
        const int hy = p / HX, hx = p % HX;
        const int gh = ty0 - 1 + hy;
        const int gw = tx0 - 1 + hx;
        const bool inb = (p < NHALO) & ((unsigned)gh < (unsigned)HH)
                                     & ((unsigned)gw < (unsigned)WW);
        const int pixh = (gh & 127) * WW + (gw & 127);   // clamped, always valid

        // UNCONDITIONAL loads on the clamped address, select-to-zero after the
        // load: keeps EXEC uniform -> one batched global load clause.
        float xv[32];
        #pragma unroll
        for (int t = 0; t < 16; ++t) {
            xv[t]      = xb[((long)(hi * 16 + t)      << 14) + pixh];
            xv[16 + t] = xb[((long)(32 + hi * 16 + t) << 14) + pixh];
        }
        v16h bf[2];
        #pragma unroll
        for (int kk = 0; kk < 2; ++kk)
            #pragma unroll
            for (int t = 0; t < 16; ++t)
                bf[kk][t] = (_Float16)(inb ? xv[kk * 16 + t] : 0.0f);

        #pragma unroll 4
        for (int mt = 0; mt < 16; ++mt) {
            v8f c = {};
            #pragma unroll
            for (int kk = 0; kk < 2; ++kk) {
                v16h a;
                #pragma unroll
                for (int t = 0; t < 16; ++t)
                    a[t] = sUeff[(mt * 16 + col) * 64 + a_kidx(t, hi, kk)];
                c = __builtin_amdgcn_wmma_f32_16x16x32_f16(false, a, false, bf[kk],
                                                           (short)0, c, false, false);
            }
            if (p < NHALO) {                     // divergent only around ds store
                _Float16* up = sU + p * 256 + mt * 16 + 8 * hi;
                #pragma unroll
                for (int t = 0; t < 8; ++t) up[t] = (_Float16)c[t];
            }
        }
    }

    __syncthreads();   // sU complete

    // ---- phase C: per-core-pixel gating weights (one lane per pixel) --------
    if (lane < 16) {
        float gv[16];
        #pragma unroll
        for (int t = 0; t < 16; ++t) gv[t] = sG[wave][t][lane];

        float gate[4], th[4], ba[4], hy4[4];
        #pragma unroll
        for (int r = 0; r < 4; ++r) {
            gate[r] = gv[r]      + gate_b[r];
            th[r]   = gv[4 + r]  + geom_b[r];
            ba[r]   = gv[8 + r]  + geom_b[4 + r];
            hy4[r]  = gv[12 + r] + geom_b[8 + r];
        }
        float mx = fmaxf(fmaxf(gate[0], gate[1]), fmaxf(gate[2], gate[3]));
        float mu[4], msum = 0.0f;
        #pragma unroll
        for (int r = 0; r < 4; ++r) { mu[r] = __expf(gate[r] - mx); msum += mu[r]; }
        float minv = 1.0f / msum;

        float wv[36], tot = 0.0f;
        #pragma unroll
        for (int r = 0; r < 4; ++r) {
            float base = softplus_f(ba[r]) + 1e-4f;
            float hyp  = softplus_f(hy4[r]) + 0.1f;
            float eh   = __expf(hyp);
            float isu  = 1.0f / (base * eh);
            float iss  = eh / base;
            float ct   = __cosf(th[r]);
            float st   = __sinf(th[r]);
            float m    = mu[r] * minv;
            #pragma unroll
            for (int s = 0; s < 9; ++s) {
                float dy = (float)(s / 3 - 1);
                float dx = (float)(s % 3 - 1);
                float pu = (ct * dx + st * dy) * isu;
                float ps = (-st * dx + ct * dy) * iss;
                float v  = m * __expf(-(pu * pu) - (ps * ps));
                wv[r * 9 + s] = v;
                tot += v;
            }
        }
        float inv = 1.0f / (tot + 1e-6f);
        float* wp = sW[wave * 16 + lane];
        #pragma unroll
        for (int i = 0; i < 36; ++i) wp[i] = wv[i] * inv;
    }

    __syncthreads();   // sW complete

    // ---- phase D: 9-point weighted gather from LDS ---------------------------
    // thread -> (core pixel q = tid>>1, output half = tid&1 -> 32 channels)
    {
        const int q    = tid >> 1;
        const int half = tid & 1;
        const int cy   = q >> 4;
        const int cx   = q & 15;

        float w36[36];
        {
            const float* wp = sW[q];
            #pragma unroll
            for (int i = 0; i < 36; ++i) w36[i] = wp[i];
        }

        float acc[32];
        #pragma unroll
        for (int j = 0; j < 32; ++j) acc[j] = pw_b[half * 32 + j];

        #pragma unroll
        for (int s = 0; s < 9; ++s) {
            const int hp = (cy + s / 3) * HX + (cx + s % 3);   // halo coords
            const v4h* up = (const v4h*)(sU + hp * 256) + half * 16;
            const float w0 = w36[s], w1 = w36[9 + s],
                        w2 = w36[18 + s], w3 = w36[27 + s];
            #pragma unroll
            for (int j = 0; j < 16; ++j) {      // 32 x ds_load_b64 per s
                v4h u4 = up[j];
                acc[j] += w0 * (float)u4[0] + w1 * (float)u4[1] +
                          w2 * (float)u4[2] + w3 * (float)u4[3];
                v4h u4b = up[16 + j];
                acc[16 + j] += w0 * (float)u4b[0] + w1 * (float)u4b[1] +
                               w2 * (float)u4b[2] + w3 * (float)u4b[3];
            }
        }

        const int pix = (ty0 + cy) * WW + tx0 + cx;
        float* op = out + ((long)b << 20) + ((long)(half * 32) << 14) + pix;
        #pragma unroll
        for (int j = 0; j < 32; ++j) op[(long)j << 14] = acc[j];
    }
}

// ---------------------------------------------------------------------------
extern "C" void kernel_launch(void* const* d_in, const int* in_sizes, int n_in,
                              void* d_out, int out_size, void* d_ws, size_t ws_size,
                              hipStream_t stream) {
    const float* x       = (const float*)d_in[0];
    const float* gate_w  = (const float*)d_in[1];
    const float* gate_b  = (const float*)d_in[2];
    const float* value_w = (const float*)d_in[3];
    const float* geom_w  = (const float*)d_in[4];
    const float* geom_b  = (const float*)d_in[5];
    const float* pw_w    = (const float*)d_in[6];
    const float* pw_b    = (const float*)d_in[7];
    float* out = (float*)d_out;

    char* ws = (char*)d_ws;                       // needs only ~66 KB scratch
    _Float16* ueff = (_Float16*)(ws + UEFF_OFF);
    _Float16* wg16 = (_Float16*)(ws + WG16_OFF);

    k_prep<<<1, 256, 0, stream>>>(gate_w, value_w, geom_w, pw_w, ueff, wg16);
    // 8 batches * 128 tiles (16x8 grid of 8x16 tiles) = 1024 blocks
    k_fused<<<1024, 256, 0, stream>>>(x, gate_b, geom_b, pw_b, ueff, wg16, out);
}
